// CapsNet_CIFAR100_12601434046771
// MI455X (gfx1250) — matrix operand order repaired and checked by reading, not compiled, not run on hardware
//
#include <hip/hip_runtime.h>
#include <math.h>

typedef __attribute__((ext_vector_type(16))) _Float16 v16h;
typedef __attribute__((ext_vector_type(8)))  _Float16 v8h;
typedef __attribute__((ext_vector_type(8)))  float    v8f;

#define DEV static __device__ __forceinline__

#if __has_builtin(__builtin_amdgcn_global_load_async_to_lds_b128) && \
    __has_builtin(__builtin_amdgcn_s_wait_asynccnt)
#define USE_ASYNC_LDS 1
// Exact parameter pointee type per hipcc diagnostic:
//   "__attribute__((__vector_size__(4 * sizeof(int)))) int __device__ *"
typedef int v4i_g __attribute__((vector_size(16)));
typedef __attribute__((address_space(1))) v4i_g as1_v4i;   // printed as "__device__"
typedef __attribute__((address_space(3))) v4i_g as3_v4i;   // printed as "__shared__"
#else
#define USE_ASYNC_LDS 0
#endif

DEV v16h frag_cat(v8h lo, v8h hi) {
  return __builtin_shufflevector(lo, hi, 0,1,2,3,4,5,6,7,8,9,10,11,12,13,14,15);
}

DEV float squash_scale(float s2) {
  // (|s|^2 / (1+|s|^2)) / sqrt(|s|^2 + eps)
  return (s2 / (1.0f + s2)) * rsqrtf(s2 + 1e-8f);
}

// A fragment (16x32 f16): lane holds row M=lane%16; lanes<16 carry K {kk..+7, kk+16..+23},
// lanes>=16 carry {kk+8..+15, kk+24..+31}. A stored [M][Kp], K contiguous.
DEV v16h loadA_frag(const _Float16* __restrict__ A, int m, int Kp, int kk, int laneHi) {
  const _Float16* pa = A + (size_t)m * Kp + kk + laneHi * 8;
  return frag_cat(*(const v8h*)pa, *(const v8h*)(pa + 16));
}

// B fragment (32x16 f16) from an LDS tile stored [n][32] (K contiguous, 64B rows):
// lane holds column N=lane%16; lanes<16 carry K {0..7, 16..23}, lanes>=16 carry {8..15, 24..31}.
DEV v16h ldsB_frag(const _Float16* base, int n, int laneHi) {
  const _Float16* p = base + n * 32 + laneHi * 8;
  return frag_cat(*(const v8h*)p, *(const v8h*)(p + 16));
}

// ---------------------------------------------------------------------------
// Packing kernels: f32 weights -> f16, K-contiguous, zero-padded K
// ---------------------------------------------------------------------------

// conv_w [256][3*81=243] -> A1 [256][256] f16 (pad 243..255 with 0)
__global__ void pack_w1_kernel(const float* __restrict__ w, _Float16* __restrict__ A1) {
  int idx = blockIdx.x * 256 + threadIdx.x;            // 256*256
  int oc = idx >> 8, kp = idx & 255;
  _Float16 v = (_Float16)0.0f;
  if (kp < 243) v = (_Float16)w[oc * 243 + kp];
  A1[idx] = v;
}

// pcap_w [256][20736] -> A2 f16 (K already mult of 32)
__global__ void pack_w2_kernel(const float* __restrict__ w, _Float16* __restrict__ A2) {
  int idx = blockIdx.x * 256 + threadIdx.x;            // 256*20736
  A2[idx] = (_Float16)w[idx];
}

// routing W [100][2048][16][8] -> straight f16 copy
__global__ void pack_wr_kernel(const float* __restrict__ w, _Float16* __restrict__ Wr) {
  int idx = blockIdx.x * 256 + threadIdx.x;            // 26,214,400
  Wr[idx] = (_Float16)w[idx];
}

// ---------------------------------------------------------------------------
// im2col kernels (B matrices stored [N][Kpad], K contiguous per position)
// ---------------------------------------------------------------------------

// x [64][3][32][32] f32 -> B1 [36864][256] f16 ; n=b*576+oy*24+ox ; k=c*81+ky*9+kx
__global__ void im2col1_kernel(const float* __restrict__ x, _Float16* __restrict__ B1) {
  int idx = blockIdx.x * 256 + threadIdx.x;            // 36864*256 = 9,437,184
  int n = idx >> 8, kp = idx & 255;
  _Float16 v = (_Float16)0.0f;
  if (kp < 243) {
    int b = n / 576, rem = n % 576, oy = rem / 24, ox = rem % 24;
    int c = kp / 81, r2 = kp % 81, ky = r2 / 9, kx = r2 % 9;
    v = (_Float16)x[((b * 3 + c) * 32 + (oy + ky)) * 32 + (ox + kx)];
  }
  B1[idx] = v;
}

// h [64][256][24][24] f16 -> B2 [4096][20736] f16 ; n=b*64+oy*8+ox (stride 2)
__global__ void im2col2_kernel(const _Float16* __restrict__ H, _Float16* __restrict__ B2) {
  int idx = blockIdx.x * 256 + threadIdx.x;            // 4096*20736 = 84,934,656
  int n = idx / 20736, k = idx % 20736;
  int b = n >> 6, yx = n & 63, oy = yx >> 3, ox = yx & 7;
  int c = k / 81, r2 = k % 81, ky = r2 / 9, kx = r2 % 9;
  B2[idx] = H[((b * 256 + c) * 24 + (2 * oy + ky)) * 24 + (2 * ox + kx)];
}

// ---------------------------------------------------------------------------
// Block GEMM with LDS-staged, double-buffered B tile.
//   Block = 256 threads (8 waves) -> 128(M) x 64(N) C tile.
//   Per k-step (K+=32): B tile 64x32 f16 (4KB) shared by all 8 waves, filled with
//   async global->LDS copies (ASYNCcnt) or load+ds_store fallback; A fragments are
//   wave-private, register-pipelined.
//   MODE 0: conv1 epilogue (bias+ReLU, f16 scatter to h[b][oc][oy][ox])
//   MODE 1: conv2 epilogue (bias, f32 store to p[b][oc][yx])
// ---------------------------------------------------------------------------
template <int KP, int NB, int MODE>
__global__ void gemm_lds_kernel(const _Float16* __restrict__ A,
                                const _Float16* __restrict__ Bg,
                                const float* __restrict__ bias,
                                void* __restrict__ Out) {
  __shared__ __align__(16) _Float16 Bt[2][64][32];
  const int tid = threadIdx.x;
  const int lane = tid & 31;
  const int wave = tid >> 5;
  const int nb = blockIdx.x % NB;
  const int mb = blockIdx.x / NB;
  const int laneHi = lane >> 4;
  const int lan16 = lane & 15;
  const int m = mb * 128 + wave * 16 + lan16;

  // cooperative fill mapping: 256 threads x 16B = 4KB tile
  const int fn = tid >> 2;                  // 0..63  (N within tile)
  const int fk = (tid & 3) * 8;             // 0,8,16,24 (K within step)
  const _Float16* gsrc = Bg + (size_t)(nb * 64 + fn) * KP + fk;

  auto fill = [&](int buf, int kk) {
#if USE_ASYNC_LDS
    __builtin_amdgcn_global_load_async_to_lds_b128(
        (as1_v4i*)(const void*)(gsrc + kk),
        (as3_v4i*)(void*)&Bt[buf][fn][fk], 0, 0);
#else
    *(v8h*)&Bt[buf][fn][fk] = *(const v8h*)(gsrc + kk);
#endif
  };
  auto waitfill = [&]() {
#if USE_ASYNC_LDS
    __builtin_amdgcn_s_wait_asynccnt(0);
#endif
    __syncthreads();
  };

  v8f acc[4] = {};
  fill(0, 0);
  waitfill();
  v16h a_cur = loadA_frag(A, m, KP, 0, laneHi);

  for (int kk = 0; kk < KP; kk += 32) {
    const int cur = (kk >> 5) & 1;
    const bool more = (kk + 32) < KP;
    v16h a_nxt = a_cur;
    if (more) {
      fill(cur ^ 1, kk + 32);               // overlaps with this step's WMMAs
      a_nxt = loadA_frag(A, m, KP, kk + 32, laneHi);
    }
#pragma unroll
    for (int f = 0; f < 4; ++f) {
      v16h bf = ldsB_frag(&Bt[cur][0][0], f * 16 + lan16, laneHi);
      acc[f] = __builtin_amdgcn_wmma_f32_16x16x32_f16(
          false, a_cur, false, bf, (short)0, acc[f], false, false);
    }
    if (more) waitfill();                   // asynccnt drain + barrier
    a_cur = a_nxt;
  }

#pragma unroll
  for (int f = 0; f < 4; ++f) {
    int n = nb * 64 + f * 16 + lan16;
#pragma unroll
    for (int j = 0; j < 8; ++j) {
      int oc = mb * 128 + wave * 16 + j + laneHi * 8;  // C/D layout: lanes>=16 hold M=8+j
      float v = acc[f][j] + bias[oc];
      if (MODE == 0) {
        v = v > 0.0f ? v : 0.0f;
        int b = n / 576, rem = n % 576, oy = rem / 24, ox = rem % 24;
        ((_Float16*)Out)[((b * 256 + oc) * 24 + oy) * 24 + ox] = (_Float16)v;
      } else {
        int b = n >> 6, yx = n & 63;
        ((float*)Out)[b * 16384 + oc * 64 + yx] = v;
      }
    }
  }
}

// ---------------------------------------------------------------------------
// Capsule squash of p rows: u[b][i][k] = squash(p[b, i*8 .. i*8+8))   (i=c*8+y, k=x)
// ---------------------------------------------------------------------------
__global__ void squash_u_kernel(const float* __restrict__ P,
                                float* __restrict__ U32,
                                _Float16* __restrict__ Uh) {
  int idx = blockIdx.x * 256 + threadIdx.x;            // 64*2048 = 131072 capsules
  const float* p = P + (size_t)idx * 8;
  float t[8], s2 = 0.0f;
#pragma unroll
  for (int k = 0; k < 8; ++k) { t[k] = p[k]; s2 += t[k] * t[k]; }
  float sc = squash_scale(s2);
#pragma unroll
  for (int k = 0; k < 8; ++k) {
    float v = t[k] * sc;
    U32[(size_t)idx * 8 + k] = v;
    Uh[(size_t)idx * 8 + k] = (_Float16)v;
  }
}

// ---------------------------------------------------------------------------
// Routing
// ---------------------------------------------------------------------------
__global__ void zero_b_kernel(float* __restrict__ Blog) {
  int idx = blockIdx.x * 256 + threadIdx.x;            // 64*100*2048
  Blog[idx] = 0.0f;
}

// softmax over o (100) for each (b,i): c[b,o,i] = softmax_o(blog[b,o,i])
__global__ void softmax_o_kernel(const float* __restrict__ Blog, float* __restrict__ Cc) {
  int idx = blockIdx.x * 256 + threadIdx.x;            // 131072
  int b = idx >> 11, i = idx & 2047;
  const float* bl = Blog + (size_t)b * 100 * 2048 + i;
  float* cr = Cc + (size_t)b * 100 * 2048 + i;
  float mx = -1e30f;
  for (int o = 0; o < 100; ++o) { float v = bl[(size_t)o * 2048]; mx = v > mx ? v : mx; }
  float s = 0.0f;
  for (int o = 0; o < 100; ++o) { float e = expf(bl[(size_t)o * 2048] - mx); cr[(size_t)o * 2048] = e; s += e; }
  float inv = 1.0f / s;
  for (int o = 0; o < 100; ++o) cr[(size_t)o * 2048] *= inv;
}

// Fused u_hat + weighted reduce:
//   S[b,o,d] = sum_{i,k} W[o,i,d,k] * ( c[b,o,i] * u[b,i,k] )
// Per block: one (o, b-tile-of-16); 4 waves split K'=16384 (K' = i*8+k),
// each wave does 128 WMMA steps (software-pipelined); LDS reduction across waves.
__global__ void route_gemm_kernel(const _Float16* __restrict__ Uh,
                                  const _Float16* __restrict__ Wr,
                                  const float* __restrict__ Cc,
                                  float* __restrict__ S) {
  __shared__ float red[4][32][8];
  const int lane = threadIdx.x & 31;
  const int wave = threadIdx.x >> 5;
  const int o = blockIdx.x >> 2;
  const int bt = blockIdx.x & 3;
  const int laneHi = lane >> 4;
  const int lan16 = lane & 15;
  const int b = bt * 16 + lan16;                       // A row (M = lane%16)
  const int d = lan16;                                 // B column
  const float* crow = Cc + ((size_t)b * 100 + o) * 2048;
  const _Float16* ub = Uh + (size_t)b * 16384;         // [2048][8]
  const _Float16* wo = Wr + (size_t)o * 2048 * 128;    // [2048][16][8]

  auto loadA = [&](int step) -> v16h {
    int iBase = (wave * 128 + step) * 4;               // kk>>3, kk = step*32
    int ia0 = iBase + laneHi;                          // lanes<16: K'{kk..+7}, hi: {+8..+15}
    int ia1 = ia0 + 2;                                 // {kk+16..+23} / {kk+24..+31}
    v8h a0 = *(const v8h*)(ub + ia0 * 8) * (_Float16)crow[ia0];
    v8h a1 = *(const v8h*)(ub + ia1 * 8) * (_Float16)crow[ia1];
    return frag_cat(a0, a1);
  };
  auto loadB = [&](int step) -> v16h {
    int iBase = (wave * 128 + step) * 4;
    int ib0 = iBase + laneHi;                          // lanes<16: K'{kk..+7, kk+16..+23}, hi: {+8..+15, +24..+31}
    v8h b0 = *(const v8h*)(wo + (ib0 * 16 + d) * 8);
    v8h b1 = *(const v8h*)(wo + ((ib0 + 2) * 16 + d) * 8);
    return frag_cat(b0, b1);
  };

  v8f acc = {};
  v16h a_cur = loadA(0);
  v16h b_cur = loadB(0);
  for (int step = 0; step < 127; ++step) {
    v16h a_nxt = loadA(step + 1);
    v16h b_nxt = loadB(step + 1);
    acc = __builtin_amdgcn_wmma_f32_16x16x32_f16(
        false, a_cur, false, b_cur, (short)0, acc, false, false);
    a_cur = a_nxt;
    b_cur = b_nxt;
  }
  acc = __builtin_amdgcn_wmma_f32_16x16x32_f16(
      false, a_cur, false, b_cur, (short)0, acc, false, false);

#pragma unroll
  for (int j = 0; j < 8; ++j) red[wave][lane][j] = acc[j];
  __syncthreads();
  if (wave == 0) {
#pragma unroll
    for (int j = 0; j < 8; ++j) {
      float v = red[0][lane][j] + red[1][lane][j] + red[2][lane][j] + red[3][lane][j];
      int bb = bt * 16 + j + laneHi * 8;               // C/D layout M mapping
      S[((size_t)bb * 100 + o) * 16 + d] = v;
    }
  }
}

// v[b,o,:] = squash(S[b,o,:]) ; written straight into d_out [64][100][16]
__global__ void squash_v_kernel(const float* __restrict__ S, float* __restrict__ V) {
  int idx = blockIdx.x * 256 + threadIdx.x;            // 6400
  if (idx >= 6400) return;
  const float* s = S + (size_t)idx * 16;
  float t[16], s2 = 0.0f;
#pragma unroll
  for (int k = 0; k < 16; ++k) { t[k] = s[k]; s2 += t[k] * t[k]; }
  float sc = squash_scale(s2);
#pragma unroll
  for (int k = 0; k < 16; ++k) V[(size_t)idx * 16 + k] = t[k] * sc;
}

// blog[b,o,i] += sum_{d,k} v[b,o,d] * W[o,i,d,k] * u[b,i,k]
__global__ void bupdate_kernel(const float* __restrict__ V,
                               const float* __restrict__ U32,
                               const _Float16* __restrict__ Wr,
                               float* __restrict__ Blog) {
  int idx = blockIdx.x * 256 + threadIdx.x;            // 64*100*2048 = 13,107,200
  int i = idx & 2047;
  int t = idx >> 11;
  int o = t % 100, b = t / 100;
  const float* v = V + ((size_t)b * 100 + o) * 16;
  const float* u = U32 + ((size_t)b * 2048 + i) * 8;
  const _Float16* w = Wr + ((size_t)o * 2048 + i) * 128;   // [16][8]
  float uf[8];
#pragma unroll
  for (int k = 0; k < 8; ++k) uf[k] = u[k];
  float delta = 0.0f;
#pragma unroll
  for (int dd = 0; dd < 16; ++dd) {
    v8h wv = *(const v8h*)(w + dd * 8);
    float dot = 0.0f;
#pragma unroll
    for (int k = 0; k < 8; ++k) dot += (float)wv[k] * uf[k];
    delta += v[dd] * dot;
  }
  Blog[((size_t)b * 100 + o) * 2048 + i] += delta;
}

// ---------------------------------------------------------------------------
// Host launcher
// ---------------------------------------------------------------------------
extern "C" void kernel_launch(void* const* d_in, const int* in_sizes, int n_in,
                              void* d_out, int out_size, void* d_ws, size_t ws_size,
                              hipStream_t stream) {
  const float* x      = (const float*)d_in[0];   // [64,3,32,32]
  const float* conv_w = (const float*)d_in[1];   // [256,3,9,9]
  const float* conv_b = (const float*)d_in[2];   // [256]
  const float* pcap_w = (const float*)d_in[3];   // [256,256,9,9]
  const float* pcap_b = (const float*)d_in[4];   // [256]
  const float* W      = (const float*)d_in[5];   // [100,2048,16,8]
  float* out = (float*)d_out;                    // [64,100,16]

  char* ws = (char*)d_ws;
  size_t off = 0;
  auto carve = [&](size_t bytes) -> char* {
    char* p = ws + off;
    off += (bytes + 255) & ~(size_t)255;
    return p;
  };
  _Float16* A1  = (_Float16*)carve((size_t)256 * 256 * 2);        // conv1 weights f16
  _Float16* A2  = (_Float16*)carve((size_t)256 * 20736 * 2);      // conv2 weights f16
  _Float16* Wr  = (_Float16*)carve((size_t)100 * 2048 * 128 * 2); // routing W f16
  _Float16* B1  = (_Float16*)carve((size_t)36864 * 256 * 2);      // im2col of x
  _Float16* Hh  = (_Float16*)carve((size_t)64 * 256 * 24 * 24 * 2); // relu conv1 out
  _Float16* B2  = (_Float16*)carve((size_t)4096 * 20736 * 2);     // im2col of h
  float*    P   = (float*)carve((size_t)64 * 16384 * 4);          // conv2 out
  float*    U32 = (float*)carve((size_t)64 * 2048 * 8 * 4);       // squashed caps f32
  _Float16* Uh  = (_Float16*)carve((size_t)64 * 2048 * 8 * 2);    // squashed caps f16
  float*    Blg = (float*)carve((size_t)64 * 100 * 2048 * 4);     // routing logits
  float*    Cc  = (float*)carve((size_t)64 * 100 * 2048 * 4);     // routing coeffs
  float*    S   = (float*)carve((size_t)64 * 100 * 16 * 4);       // pre-squash v
  (void)ws_size; (void)in_sizes; (void)n_in; (void)out_size;

  // ---- pack weights ----
  pack_w1_kernel<<<256, 256, 0, stream>>>(conv_w, A1);
  pack_w2_kernel<<<20736, 256, 0, stream>>>(pcap_w, A2);
  pack_wr_kernel<<<102400, 256, 0, stream>>>(W, Wr);

  // ---- conv stem: M=256, N=36864, K=256 ----
  im2col1_kernel<<<36864, 256, 0, stream>>>(x, B1);
  gemm_lds_kernel<256, 576, 0><<<1152, 256, 0, stream>>>(A1, B1, conv_b, (void*)Hh);

  // ---- primary caps: M=256, N=4096, K=20736 ----
  im2col2_kernel<<<331776, 256, 0, stream>>>(Hh, B2);
  gemm_lds_kernel<20736, 64, 1><<<128, 256, 0, stream>>>(A2, B2, pcap_b, (void*)P);

  squash_u_kernel<<<512, 256, 0, stream>>>(P, U32, Uh);

  // ---- dynamic routing (3 iterations) ----
  zero_b_kernel<<<51200, 256, 0, stream>>>(Blg);
  for (int it = 0; it < 3; ++it) {
    softmax_o_kernel<<<512, 256, 0, stream>>>(Blg, Cc);
    route_gemm_kernel<<<400, 128, 0, stream>>>(Uh, Wr, Cc, S);       // 100 o x 4 b-tiles
    squash_v_kernel<<<25, 256, 0, stream>>>(S, out);
    if (it < 2) bupdate_kernel<<<51200, 256, 0, stream>>>(out, U32, Wr, Blg);
  }
}
